// DHSM_37014028157380
// MI455X (gfx1250) — compile-verified, matrix operand order checked by reading, not run on hardware
//
#include <hip/hip_runtime.h>
#include <hip/hip_bf16.h>
#include <math.h>

typedef _Float16 half_t;
typedef __attribute__((ext_vector_type(16))) _Float16 v16h;
typedef __attribute__((ext_vector_type(8)))  float    v8f;

// model dims
constexpr int Bn = 4, Sn = 512, Hn = 1024, Nn = 128, Ln = 6, Vn = 32000;
constexpr int NHn = 8, FFn = 4096, DHn = Hn / NHn;     // 128
constexpr int Mn = Bn * Sn;                            // 2048 tokens

// GEMM tiling: 128x128 block, K-depth 64 per LDS stage (2 WMMA k-steps)
constexpr int BMt = 128, BNt = 128, BKt = 64, LDSK = BKt + 8; // 72 halves -> 144B rows (16B aligned)

union Frag { v16h h; float4 q[2]; };

// async copy: 16B global -> LDS through the ASYNCcnt engine (CDNA5)
__device__ __forceinline__ void async_b128(const half_t* gp, half_t* lp) {
  unsigned lds_off = (unsigned)(size_t)lp;            // low 32b of generic shared ptr = LDS offset
  asm volatile("global_load_async_to_lds_b128 %0, %1, off"
               :: "v"(lds_off), "v"(gp) : "memory");
}
__device__ __forceinline__ void wait_async0() {
  asm volatile("s_wait_asynccnt 0x0" ::: "memory");
}

// ---------------------------------------------------------------------------
// f32 -> f16 conversion (weights / activations)
// ---------------------------------------------------------------------------
__global__ void k_cvt(const float* __restrict__ s, half_t* __restrict__ d, long n) {
  long i = ((long)blockIdx.x * blockDim.x + threadIdx.x) * 4;
  if (i + 3 < n) {
    float4 v = *reinterpret_cast<const float4*>(s + i);
    d[i + 0] = (half_t)v.x; d[i + 1] = (half_t)v.y;
    d[i + 2] = (half_t)v.z; d[i + 3] = (half_t)v.w;
  }
}

// ---------------------------------------------------------------------------
// embedding + positional add
// ---------------------------------------------------------------------------
__global__ void k_embed(const int* __restrict__ ids, const float* __restrict__ emb,
                        const float* __restrict__ pos, float* __restrict__ x,
                        half_t* __restrict__ xh) {
  int tok = blockIdx.x;
  int s   = tok % Sn;
  long id = ids[tok];
  for (int h = threadIdx.x; h < Hn; h += 256) {
    float v = emb[id * Hn + h] + pos[(long)s * Hn + h];
    x [(long)tok * Hn + h] = v;
    xh[(long)tok * Hn + h] = (half_t)v;
  }
}

// ---------------------------------------------------------------------------
// Generic WMMA GEMM: C[z][m,n] = A[z][m,:] dot W[z][n,:]  (+bias, +act)
//   A: K-contiguous, leading-dim lda.
//   W: element (n,k) at n*sWn + k*sWk. VECW => sWk==1 (async 16B staging).
//   batch z: base += (z/zdiv)*zHi + (z%zdiv)*zLo  per operand.
//   ACT: 0 none, 1 exact GELU, 2 sigmoid. Optional f16 mirror store Ch.
//   Double-buffered LDS (72KB); tiles staged via GLOBAL_LOAD_ASYNC_TO_LDS;
//   16 WMMAs per wave between barriers.
// ---------------------------------------------------------------------------
template<bool VECW, int ACT>
__global__ __launch_bounds__(256)
void k_gemm(const half_t* __restrict__ A, long lda, long zHiA, long zLoA,
            const half_t* __restrict__ W, long sWn, long sWk, long zHiW, long zLoW,
            const float* __restrict__ bias,
            float* __restrict__ C, long ldc, long zHiC, long zLoC,
            half_t* __restrict__ Ch, long ldch,
            int K, int zdiv) {
  __shared__ __align__(16) half_t As[2][BMt * LDSK];
  __shared__ __align__(16) half_t Ws[2][BNt * LDSK];

  const int tid  = threadIdx.x;
  const int lane = tid & 31;
  const int wave = tid >> 5;
  const int wm   = wave >> 1;   // 0..3  (M)
  const int wn   = wave & 1;    // 0..1  (N)

  const int z = blockIdx.z;
  const long zb = z / zdiv, zr = z % zdiv;
  A += zb * zHiA + zr * zLoA;
  W += zb * zHiW + zr * zLoW;
  const long coff = zb * zHiC + zr * zLoC;
  C += coff;
  if (Ch) Ch += coff;

  const int m0 = blockIdx.y * BMt;
  const int n0 = blockIdx.x * BNt;

  // staging: 128x64 halves = 1024 16B chunks, 4 per thread per matrix
  auto stageA = [&](int buf, int k0) {
#pragma unroll
    for (int i = 0; i < 4; ++i) {
      int c = tid * 4 + i;
      int r = c >> 3, ko = (c & 7) * 8;
      async_b128(A + (long)(m0 + r) * lda + (k0 + ko), &As[buf][r * LDSK + ko]);
    }
  };
  auto stageW = [&](int buf, int k0) {
    if (VECW) {
#pragma unroll
      for (int i = 0; i < 4; ++i) {
        int c = tid * 4 + i;
        int r = c >> 3, ko = (c & 7) * 8;
        async_b128(W + (long)(n0 + r) * sWn + (k0 + ko), &Ws[buf][r * LDSK + ko]);
      }
    } else {
#pragma unroll
      for (int i = 0; i < 32; ++i) {      // strided gather (att @ V): synchronous b16 stores
        int e = tid * 32 + i;
        int n = e >> 6, k = e & 63;
        Ws[buf][n * LDSK + k] = W[(long)(n0 + n) * sWn + (long)(k0 + k) * sWk];
      }
    }
  };

  v8f acc[2][4] = {};

  stageA(0, 0);
  stageW(0, 0);
  wait_async0();
  __syncthreads();

  int cur = 0;
  for (int k0 = 0; k0 < K; k0 += BKt) {
    const int  nxt  = cur ^ 1;
    const bool more = (k0 + BKt) < K;
    if (more) {                 // kick off next tile while the XDL pipe works on this one
      stageA(nxt, k0 + BKt);
      stageW(nxt, k0 + BKt);
    }
    if (k0 + 2 * BKt < K)       // L2 warm-up two tiles ahead
      __builtin_prefetch(A + (long)(m0 + (tid >> 1)) * lda + (k0 + 2 * BKt), 0, 1);

    // ---- two WMMA k-steps per staged tile ----
    const int kb = (lane < 16) ? 0 : 8;
    const int rl = lane & 15;
#pragma unroll
    for (int kk = 0; kk < BKt; kk += 32) {
      Frag a[2], b[4];
#pragma unroll
      for (int mi = 0; mi < 2; ++mi) {
        int row = wm * 32 + mi * 16 + rl;
        a[mi].q[0] = *reinterpret_cast<const float4*>(&As[cur][row * LDSK + kk + kb]);
        a[mi].q[1] = *reinterpret_cast<const float4*>(&As[cur][row * LDSK + kk + kb + 16]);
      }
#pragma unroll
      for (int ni = 0; ni < 4; ++ni) {
        int col = wn * 64 + ni * 16 + rl;
        b[ni].q[0] = *reinterpret_cast<const float4*>(&Ws[cur][col * LDSK + kk + kb]);
        b[ni].q[1] = *reinterpret_cast<const float4*>(&Ws[cur][col * LDSK + kk + kb + 16]);
      }
#pragma unroll
      for (int mi = 0; mi < 2; ++mi)
#pragma unroll
        for (int ni = 0; ni < 4; ++ni)
          acc[mi][ni] = __builtin_amdgcn_wmma_f32_16x16x32_f16(
              false, a[mi].h, false, b[ni].h, (short)0, acc[mi][ni], false, false);
    }

    if (more) wait_async0();    // next tile landed in LDS
    __syncthreads();            // all waves done reading cur / writing nxt
    cur = nxt;
  }

  // ---- epilogue: bias + activation, fp32 store (+ optional f16 mirror) ----
  const int rl  = lane & 15;
  const int rhi = (lane >> 4) * 8;
#pragma unroll
  for (int mi = 0; mi < 2; ++mi) {
#pragma unroll
    for (int ni = 0; ni < 4; ++ni) {
#pragma unroll
      for (int v = 0; v < 8; ++v) {
        int row = m0 + wm * 32 + mi * 16 + rhi + v;
        int col = n0 + wn * 64 + ni * 16 + rl;
        float cv = acc[mi][ni][v];
        if (bias) cv += bias[col];
        if (ACT == 1) cv = 0.5f * cv * (1.0f + erff(cv * 0.70710678118654752f));
        else if (ACT == 2) cv = 1.0f / (1.0f + __expf(-cv));
        C[(long)row * ldc + col] = cv;
        if (Ch) Ch[(long)row * ldch + col] = (half_t)cv;
      }
    }
  }
}

template<bool V, int A_>
static void launch_gemm(hipStream_t stream,
                        const half_t* A, long lda, long zHiA, long zLoA,
                        const half_t* W, long sWn, long sWk, long zHiW, long zLoW,
                        const float* bias, float* C, long ldc, long zHiC, long zLoC,
                        half_t* Ch, long ldch, int M, int N, int K, int Z, int zdiv) {
  dim3 g(N / BNt, M / BMt, Z);
  k_gemm<V, A_><<<g, dim3(256), 0, stream>>>(A, lda, zHiA, zLoA, W, sWn, sWk, zHiW, zLoW,
                                             bias, C, ldc, zHiC, zLoC, Ch, ldch, K, zdiv);
}

// ---------------------------------------------------------------------------
// SSM selective scan: st = scan(A*s + gate*bp) over S; one thread per (b,n)
// ---------------------------------------------------------------------------
__global__ void k_scan(const float* __restrict__ gate, const float* __restrict__ bp,
                       const float* __restrict__ A_log, half_t* __restrict__ st_h) {
  int idx = blockIdx.x * blockDim.x + threadIdx.x;
  if (idx >= Bn * Nn) return;
  int b = idx / Nn, n = idx % Nn;
  float A = __expf(A_log[n]);
  float s = 0.0f;
  long o = (long)b * Sn * Nn + n;
  for (int t = 0; t < Sn; ++t, o += Nn) {
    s = A * s + gate[o] * bp[o];
    st_h[o] = (half_t)s;
  }
}

// y += (Dp[h] + 1) * x   (in place, y = WC-projection output)
__global__ void k_dresid(float* __restrict__ y, const float* __restrict__ x,
                         const float* __restrict__ Dp) {
  long i = (long)blockIdx.x * blockDim.x + threadIdx.x;
  int h = (int)(i % Hn);
  y[i] += (Dp[h] + 1.0f) * x[i];
}

// ---------------------------------------------------------------------------
// LayerNorm over H=1024: out = scale * (LN(a (+ r)) * g + b); optional f16 copy
// ---------------------------------------------------------------------------
__global__ void k_ln(const float* __restrict__ a, const float* __restrict__ r,
                     const float* __restrict__ g, const float* __restrict__ b,
                     float* __restrict__ out, half_t* __restrict__ outh, float scale) {
  int row = blockIdx.x;
  const float* pa = a + (long)row * Hn;
  const float* pr = r ? r + (long)row * Hn : nullptr;
  __shared__ float red[8];
  float v[4];
  float sum = 0.0f;
#pragma unroll
  for (int j = 0; j < 4; ++j) {
    int h = threadIdx.x + j * 256;
    v[j] = pa[h] + (pr ? pr[h] : 0.0f);
    sum += v[j];
  }
  for (int m = 16; m > 0; m >>= 1) sum += __shfl_xor(sum, m, 32);
  int wv = threadIdx.x >> 5, ln = threadIdx.x & 31;
  if (ln == 0) red[wv] = sum;
  __syncthreads();
  float tot = 0.0f;
#pragma unroll
  for (int w = 0; w < 8; ++w) tot += red[w];
  float mu = tot * (1.0f / Hn);
  __syncthreads();
  float var = 0.0f;
#pragma unroll
  for (int j = 0; j < 4; ++j) { float d = v[j] - mu; var += d * d; }
  for (int m = 16; m > 0; m >>= 1) var += __shfl_xor(var, m, 32);
  if (ln == 0) red[wv] = var;
  __syncthreads();
  tot = 0.0f;
#pragma unroll
  for (int w = 0; w < 8; ++w) tot += red[w];
  float rs = rsqrtf(tot * (1.0f / Hn) + 1e-5f);
#pragma unroll
  for (int j = 0; j < 4; ++j) {
    int h = threadIdx.x + j * 256;
    float o = scale * ((v[j] - mu) * rs * g[h] + b[h]);
    out[(long)row * Hn + h] = o;
    if (outh) outh[(long)row * Hn + h] = (half_t)o;
  }
}

// ---------------------------------------------------------------------------
// row softmax over S=512 with 1/sqrt(DH) scaling, f16 output for att@V GEMM
// ---------------------------------------------------------------------------
__global__ void k_softmax(const float* __restrict__ sc, half_t* __restrict__ att, float scale) {
  long row = blockIdx.x;
  const float* p = sc + row * Sn;
  __shared__ float red[8];
  int t = threadIdx.x, wv = t >> 5, ln = t & 31;
  float v0 = p[t] * scale, v1 = p[t + 256] * scale;
  float mx = fmaxf(v0, v1);
  for (int m = 16; m > 0; m >>= 1) mx = fmaxf(mx, __shfl_xor(mx, m, 32));
  if (ln == 0) red[wv] = mx;
  __syncthreads();
  float MX = red[0];
#pragma unroll
  for (int w = 1; w < 8; ++w) MX = fmaxf(MX, red[w]);
  __syncthreads();
  float e0 = __expf(v0 - MX), e1 = __expf(v1 - MX);
  float sum = e0 + e1;
  for (int m = 16; m > 0; m >>= 1) sum += __shfl_xor(sum, m, 32);
  if (ln == 0) red[wv] = sum;
  __syncthreads();
  float S = 0.0f;
#pragma unroll
  for (int w = 0; w < 8; ++w) S += red[w];
  float inv = 1.0f / S;
  att[row * Sn + t]       = (half_t)(e0 * inv);
  att[row * Sn + t + 256] = (half_t)(e1 * inv);
}

// ---------------------------------------------------------------------------
extern "C" void kernel_launch(void* const* d_in, const int* in_sizes, int n_in,
                              void* d_out, int out_size, void* d_ws, size_t ws_size,
                              hipStream_t stream) {
  (void)in_sizes; (void)n_in; (void)out_size; (void)ws_size;

  const int*   ids  = (const int*)  d_in[0];
  const float* emb  = (const float*)d_in[1];
  const float* pos  = (const float*)d_in[2];
  const float* Alog = (const float*)d_in[3];
  const float* WB   = (const float*)d_in[4];
  const float* WC   = (const float*)d_in[5];
  const float* Dp   = (const float*)d_in[6];
  const float* Wg   = (const float*)d_in[7];
  const float* bg   = (const float*)d_in[8];
  const float* Wout = (const float*)d_in[9];
  const float* bout = (const float*)d_in[10];
  const float* gs   = (const float*)d_in[11];
  const float* bs   = (const float*)d_in[12];
  const float* Wqkv = (const float*)d_in[13];
  const float* bqkv = (const float*)d_in[14];
  const float* Wo   = (const float*)d_in[15];
  const float* bo   = (const float*)d_in[16];
  const float* W1   = (const float*)d_in[17];
  const float* b1   = (const float*)d_in[18];
  const float* W2   = (const float*)d_in[19];
  const float* b2   = (const float*)d_in[20];
  const float* g1   = (const float*)d_in[21];
  const float* bn1  = (const float*)d_in[22];
  const float* g2   = (const float*)d_in[23];
  const float* bn2  = (const float*)d_in[24];
  const float* g3   = (const float*)d_in[25];
  const float* bn3  = (const float*)d_in[26];
  const float* gf   = (const float*)d_in[27];
  const float* bf   = (const float*)d_in[28];

  // ---- workspace bump allocator ----
  char* base = (char*)d_ws;
  size_t off = 0;
  auto alloc = [&](size_t bytes) -> void* {
    off = (off + 255) & ~(size_t)255;
    void* p = base + off;
    off += bytes;
    return p;
  };
  const size_t MH = (size_t)Mn * Hn;

  half_t* emb_h   = (half_t*)alloc((size_t)Vn * Hn * 2);
  half_t* wscr    = (half_t*)alloc((size_t)FFn * Hn * 2);    // shared weight-f16 slot
  float*  x       = (float*) alloc(MH * 4);
  half_t* xh      = (half_t*)alloc(MH * 2);
  float*  gate    = (float*) alloc((size_t)Mn * Nn * 4);
  float*  bp      = (float*) alloc((size_t)Mn * Nn * 4);
  half_t* st_h    = (half_t*)alloc((size_t)Mn * Nn * 2);
  float*  y1      = (float*) alloc(MH * 4);
  float*  lns     = (float*) alloc(MH * 4);
  half_t* lnsh    = (half_t*)alloc(MH * 2);
  float*  xs      = (float*) alloc(MH * 4);
  float*  x2      = (float*) alloc(MH * 4);
  half_t* x2h     = (half_t*)alloc(MH * 2);
  float*  bufA    = (float*) alloc((size_t)Mn * FFn * 4);    // qkv / ff1
  half_t* bufAh   = (half_t*)alloc((size_t)Mn * FFn * 2);
  float*  scores  = (float*) alloc((size_t)Bn * NHn * Sn * Sn * 4);
  half_t* att_h   = (half_t*)alloc((size_t)Bn * NHn * Sn * Sn * 2);
  float*  attout  = (float*) alloc(MH * 4);
  half_t* attouth = (half_t*)alloc(MH * 2);
  float*  ao      = (float*) alloc(MH * 4);
  float*  x3      = (float*) alloc(MH * 4);
  half_t* x3h     = (half_t*)alloc(MH * 2);
  float*  ffout   = (float*) alloc(MH * 4);
  float*  xf      = (float*) alloc(MH * 4);
  half_t* xfh     = (half_t*)alloc(MH * 2);

  auto cvt = [&](const float* s, half_t* d, long n) {
    k_cvt<<<dim3((unsigned)((n / 4 + 255) / 256)), dim3(256), 0, stream>>>(s, d, n);
  };

  // embedding table to f16 (lm-head operand), token embed + pos
  cvt(emb, emb_h, (long)Vn * Hn);
  k_embed<<<dim3(Mn), dim3(256), 0, stream>>>(ids, emb, pos, x, xh);

  for (int i = 0; i < Ln; ++i) {
    // ---- SSM block ----
    cvt(Wg + (size_t)i * Nn * Hn, wscr, (long)Nn * Hn);
    launch_gemm<true, 2>(stream, xh, Hn, 0, 0, wscr, Hn, 1, 0, 0,
                         bg + (size_t)i * Nn, gate, Nn, 0, 0, nullptr, 0,
                         Mn, Nn, Hn, 1, 1);
    cvt(WB + (size_t)i * Nn * Hn, wscr, (long)Nn * Hn);
    launch_gemm<true, 0>(stream, xh, Hn, 0, 0, wscr, Hn, 1, 0, 0,
                         nullptr, bp, Nn, 0, 0, nullptr, 0,
                         Mn, Nn, Hn, 1, 1);
    k_scan<<<dim3(2), dim3(256), 0, stream>>>(gate, bp, Alog + (size_t)i * Nn, st_h);
    cvt(WC + (size_t)i * Hn * Nn, wscr, (long)Hn * Nn);
    launch_gemm<true, 0>(stream, st_h, Nn, 0, 0, wscr, Nn, 1, 0, 0,
                         nullptr, y1, Hn, 0, 0, nullptr, 0,
                         Mn, Hn, Nn, 1, 1);
    k_dresid<<<dim3((unsigned)(MH / 256)), dim3(256), 0, stream>>>(y1, x, Dp + (size_t)i * Hn);
    k_ln<<<dim3(Mn), dim3(256), 0, stream>>>(y1, nullptr, gs + (size_t)i * Hn, bs + (size_t)i * Hn,
                                             lns, lnsh, 1.0f);
    cvt(Wout + (size_t)i * Hn * Hn, wscr, (long)Hn * Hn);
    launch_gemm<true, 0>(stream, lnsh, Hn, 0, 0, wscr, Hn, 1, 0, 0,
                         bout + (size_t)i * Hn, xs, Hn, 0, 0, nullptr, 0,
                         Mn, Hn, Hn, 1, 1);
    k_ln<<<dim3(Mn), dim3(256), 0, stream>>>(xs, x, g1 + (size_t)i * Hn, bn1 + (size_t)i * Hn,
                                             x2, x2h, 1.0f);

    if (i & 1) {
      // ---- attention ----
      cvt(Wqkv + (size_t)i * 3 * Hn * Hn, wscr, (long)3 * Hn * Hn);
      launch_gemm<true, 0>(stream, x2h, Hn, 0, 0, wscr, Hn, 1, 0, 0,
                           bqkv + (size_t)i * 3 * Hn, bufA, 3 * Hn, 0, 0, bufAh, 3 * Hn,
                           Mn, 3 * Hn, Hn, 1, 1);
      // scores[z] = Q[z] @ K[z]^T   (z = b*NH + h)
      launch_gemm<true, 0>(stream,
                           bufAh, 3 * Hn, (long)Sn * 3 * Hn, DHn,
                           bufAh + Hn, 3 * Hn, 1, (long)Sn * 3 * Hn, DHn,
                           nullptr,
                           scores, Sn, (long)NHn * Sn * Sn, (long)Sn * Sn,
                           nullptr, 0,
                           Sn, Sn, DHn, Bn * NHn, NHn);
      k_softmax<<<dim3(Bn * NHn * Sn), dim3(256), 0, stream>>>(scores, att_h, 0.08838834764831845f);
      // attout[z] = att[z] @ V[z]   (V read transposed: sWn=1, sWk=3H)
      launch_gemm<false, 0>(stream,
                            att_h, Sn, (long)NHn * Sn * Sn, (long)Sn * Sn,
                            bufAh + 2 * Hn, 1, 3 * Hn, (long)Sn * 3 * Hn, DHn,
                            nullptr,
                            attout, Hn, (long)Sn * Hn, DHn,
                            attouth, Hn,
                            Sn, DHn, Sn, Bn * NHn, NHn);
      cvt(Wo + (size_t)i * Hn * Hn, wscr, (long)Hn * Hn);
      launch_gemm<true, 0>(stream, attouth, Hn, 0, 0, wscr, Hn, 1, 0, 0,
                           bo + (size_t)i * Hn, ao, Hn, 0, 0, nullptr, 0,
                           Mn, Hn, Hn, 1, 1);
      // x3 = 2 * LN(x2 + ao)   (folds the HierarchicalMemory 2x)
      k_ln<<<dim3(Mn), dim3(256), 0, stream>>>(ao, x2, g2 + (size_t)i * Hn, bn2 + (size_t)i * Hn,
                                               x3, x3h, 2.0f);
    } else {
      k_ln<<<dim3(Mn), dim3(256), 0, stream>>>(x2, nullptr, g2 + (size_t)i * Hn, bn2 + (size_t)i * Hn,
                                               x3, x3h, 2.0f);
    }

    // ---- FFN ----
    cvt(W1 + (size_t)i * FFn * Hn, wscr, (long)FFn * Hn);
    launch_gemm<true, 1>(stream, x3h, Hn, 0, 0, wscr, Hn, 1, 0, 0,
                         b1 + (size_t)i * FFn, bufA, FFn, 0, 0, bufAh, FFn,
                         Mn, FFn, Hn, 1, 1);
    cvt(W2 + (size_t)i * Hn * FFn, wscr, (long)Hn * FFn);
    launch_gemm<true, 0>(stream, bufAh, FFn, 0, 0, wscr, FFn, 1, 0, 0,
                         b2 + (size_t)i * Hn, ffout, Hn, 0, 0, nullptr, 0,
                         Mn, Hn, FFn, 1, 1);
    // x = LN(x3 + ffout)  -> next-layer input (fp32 + f16)
    k_ln<<<dim3(Mn), dim3(256), 0, stream>>>(ffout, x3, g3 + (size_t)i * Hn, bn3 + (size_t)i * Hn,
                                             x, xh, 1.0f);
  }

  // ---- final LN + tied lm-head: logits = LN(x) @ emb^T ----
  k_ln<<<dim3(Mn), dim3(256), 0, stream>>>(x, nullptr, gf, bf, xf, xfh, 1.0f);
  launch_gemm<true, 0>(stream, xfh, Hn, 0, 0, emb_h, Hn, 1, 0, 0,
                       nullptr, (float*)d_out, Vn, 0, 0, nullptr, 0,
                       Mn, Vn, Hn, 1, 1);
}
